// Block_10548439679513
// MI455X (gfx1250) — matrix operand order repaired and checked by reading, not compiled, not run on hardware
//
#include <hip/hip_runtime.h>
#include <stdint.h>

#define B_   4
#define T_   2048
#define C_   1024
#define H_   16
#define D_   64
#define DFF_ 4096
#define BT_  (B_*T_)

typedef __attribute__((ext_vector_type(16))) __bf16 v16bf;
typedef __attribute__((ext_vector_type(8)))  float v8f;
typedef __attribute__((ext_vector_type(4)))  int   v4i_t;

union Frag {
  v16bf v;
  unsigned short s[16];
  uint4 q[2];
};

__device__ __forceinline__ unsigned short f2bf(float f) {
  union { float f; unsigned int u; } c; c.f = f;
  unsigned int u = c.u;
  unsigned int r = (u + 0x7FFFu + ((u >> 16) & 1u)) >> 16;
  return (unsigned short)r;
}

// --------------------------------------------------------------------------
// CDNA5 async global->LDS copy (ASYNCcnt path), with safe fallback.
// Builtin signature (from clang diagnostic): (v4i32* AS1, v4i32* AS3, Ii, Ii)
// --------------------------------------------------------------------------
#if __has_builtin(__builtin_amdgcn_global_load_async_to_lds_b128)
#define HAVE_ASYNC_LDS 1
__device__ __forceinline__ void async_copy16(const void* g, void* l) {
  __builtin_amdgcn_global_load_async_to_lds_b128(
      (__attribute__((address_space(1))) v4i_t*)(g),
      (__attribute__((address_space(3))) v4i_t*)(l), 0, 0);
}
#else
#define HAVE_ASYNC_LDS 0
#endif

__device__ __forceinline__ void wait_async0() {
#if __has_builtin(__builtin_amdgcn_s_wait_asynccnt)
  __builtin_amdgcn_s_wait_asynccnt(0);
#else
  asm volatile("s_wait_asynccnt 0x0" ::: "memory");
#endif
}

// ---------------------------------------------------------------------------
// Transpose f32 [R,Cc] -> bf16 [Cc,R]  (weights -> Bt layout for WMMA B-frags)
// ---------------------------------------------------------------------------
__global__ __launch_bounds__(256) void transpose_bf16_kernel(
    const float* __restrict__ in, unsigned short* __restrict__ out, int R, int Cc) {
  int idx = blockIdx.x * 256 + threadIdx.x;
  if (idx < R * Cc) {
    int r = idx / Cc, c = idx % Cc;
    out[(size_t)c * R + r] = f2bf(in[idx]);
  }
}

// ---------------------------------------------------------------------------
// LayerNorm * mask -> bf16 rows (one block per row of C_ elements)
// ---------------------------------------------------------------------------
__global__ __launch_bounds__(256) void ln_kernel(
    const float* __restrict__ x, const float* __restrict__ w,
    const float* __restrict__ bia, const unsigned char* __restrict__ mask,
    unsigned short* __restrict__ out) {
  int row = blockIdx.x;
  const float* xr = x + (size_t)row * C_;
  __shared__ float red[256];
  float s = 0.f;
  for (int c = threadIdx.x; c < C_; c += 256) s += xr[c];
  red[threadIdx.x] = s; __syncthreads();
  for (int off = 128; off > 0; off >>= 1) {
    if (threadIdx.x < off) red[threadIdx.x] += red[threadIdx.x + off];
    __syncthreads();
  }
  float mu = red[0] * (1.0f / C_);
  __syncthreads();
  float vs = 0.f;
  for (int c = threadIdx.x; c < C_; c += 256) { float d = xr[c] - mu; vs += d * d; }
  red[threadIdx.x] = vs; __syncthreads();
  for (int off = 128; off > 0; off >>= 1) {
    if (threadIdx.x < off) red[threadIdx.x] += red[threadIdx.x + off];
    __syncthreads();
  }
  float rstd = rsqrtf(red[0] * (1.0f / C_) + 1e-5f);
  float mm = mask[row] ? 1.f : 0.f;
  for (int c = threadIdx.x; c < C_; c += 256)
    out[(size_t)row * C_ + c] = f2bf(((xr[c] - mu) * rstd * w[c] + bia[c]) * mm);
}

// ---------------------------------------------------------------------------
// WMMA GEMM: A bf16 [M,K] row-major, Bt bf16 [N,K] row-major, f32 accumulate.
// Block = 8 waves = 128 rows x 64 cols. B tile (64x32 = 4KB) staged into LDS
// once per block per k-step via async global->LDS b128 copies (double
// buffered); A fragment register-prefetched one k-step ahead.
// ---------------------------------------------------------------------------
#define M_QK   0   // bf16 out scattered to [B,H,T,d]
#define M_VT   1   // bf16 out scattered to [B,H,d,T]
#define M_PROJ 2   // f32 out = acc + bias[col] + resid[row,col]
#define M_RELU 3   // bf16 out = relu(acc + bias[col])
#define M_FF2  4   // f32 out = (acc + bias[col] + resid[row,col]) * mask[row]

template<int MODE>
__global__ __launch_bounds__(256) void gemm_bf16_kernel(
    const unsigned short* __restrict__ A, const unsigned short* __restrict__ Bt,
    int M, int N, int K,
    float* __restrict__ outF, unsigned short* __restrict__ outB,
    const float* __restrict__ bias, const float* __restrict__ resid,
    const unsigned char* __restrict__ mask) {
  int tid = threadIdx.x;
  int wv = tid >> 5, lane = tid & 31;
  int n  = lane & 15, hh = lane >> 4;          // hh = K-half selector
  int rowBase = (blockIdx.y * 8 + wv) * 16;
  int colBase = blockIdx.x * 64;

  v8f acc[4] = {};
  const unsigned short* arow = A + (size_t)(rowBase + n) * K;  // m = lane&15

#if HAVE_ASYNC_LDS
  __shared__ __align__(16) unsigned short Btile[2][64][32];
  int brow16 = tid >> 1, bhalf = tid & 1;      // 256 x 16B copies: rows 0..127, 8 cols at bhalf*16
  const unsigned short* bsrc = Bt + (size_t)(colBase + brow16) * K + bhalf * 16;

  // stage k0 = 0
  async_copy16(bsrc, &Btile[0][brow16][bhalf * 16]);
  wait_async0();
  __syncthreads();

  Frag a;
  a.q[0] = *(const uint4*)(arow + hh * 16);
  a.q[1] = *(const uint4*)(arow + 8 + hh * 16);

  for (int k0 = 0; k0 < K; k0 += 32) {
    int cur = (k0 >> 5) & 1;
    bool more = (k0 + 32 < K);
    if (more)  // stage next B tile into the other buffer (overlaps WMMAs)
      async_copy16(bsrc + k0 + 32, &Btile[cur ^ 1][brow16][bhalf * 16]);
    Frag anext;
    if (more) {  // prefetch next A fragment into registers
      anext.q[0] = *(const uint4*)(arow + k0 + 32 + hh * 16);
      anext.q[1] = *(const uint4*)(arow + k0 + 40 + hh * 16);
    }
#pragma unroll
    for (int j = 0; j < 4; ++j) {
      Frag b;
      const unsigned short* brow = &Btile[cur][j * 16 + n][hh * 16];
      b.q[0] = ((const uint4*)brow)[0];
      b.q[1] = ((const uint4*)brow)[1];
      acc[j] = __builtin_amdgcn_wmma_f32_16x16x32_bf16(
          false, a.v, false, b.v, (short)0, acc[j], false, false);
    }
    if (more) {
      a = anext;
      wait_async0();
    }
    __syncthreads();
  }
#else
  for (int k0 = 0; k0 < K; k0 += 32) {
    Frag a;
    a.q[0] = *(const uint4*)(arow + k0 + hh * 16);
    a.q[1] = *(const uint4*)(arow + k0 + 8 + hh * 16);
#pragma unroll
    for (int j = 0; j < 4; ++j) {
      Frag b;
      const unsigned short* brow =
          Bt + (size_t)(colBase + j * 16 + n) * K + k0 + hh * 16;
      b.q[0] = ((const uint4*)brow)[0];
      b.q[1] = ((const uint4*)brow)[1];
      acc[j] = __builtin_amdgcn_wmma_f32_16x16x32_bf16(
          false, a.v, false, b.v, (short)0, acc[j], false, false);
    }
  }
#endif

#pragma unroll
  for (int j = 0; j < 4; ++j) {
    int col = colBase + j * 16 + n;
#pragma unroll
    for (int i = 0; i < 8; ++i) {
      int row = rowBase + i + 8 * hh;
      float val = acc[j][i];
      if (MODE == M_QK) {
        int b = row / T_, t = row % T_;
        int head = col >> 6, dd = col & 63;
        outB[(((size_t)(b * H_ + head) * T_ + t) << 6) + dd] = f2bf(val);
      } else if (MODE == M_VT) {
        int b = row / T_, t = row % T_;
        int head = col >> 6, dd = col & 63;
        outB[((size_t)(b * H_ + head) * D_ + dd) * T_ + t] = f2bf(val);
      } else if (MODE == M_PROJ) {
        outF[(size_t)row * N + col] = val + bias[col] + resid[(size_t)row * N + col];
      } else if (MODE == M_RELU) {
        float r = val + bias[col];
        outB[(size_t)row * N + col] = f2bf(r > 0.f ? r : 0.f);
      } else { // M_FF2
        float mm = mask[row] ? 1.f : 0.f;
        outF[(size_t)row * N + col] =
            (val + bias[col] + resid[(size_t)row * N + col]) * mm;
      }
    }
  }
}

// ---------------------------------------------------------------------------
// Flash attention: one wave per (b, head, 16-row q-tile).
// q,k: bf16 [B,H,T,64]; vt: bf16 [B,H,64,T]; out: bf16 [B,T,C]
// ---------------------------------------------------------------------------
__global__ __launch_bounds__(256) void attn_kernel(
    const unsigned short* __restrict__ q, const unsigned short* __restrict__ k,
    const unsigned short* __restrict__ vt, const unsigned char* __restrict__ kpm,
    unsigned short* __restrict__ attn_out) {
  __shared__ float sc[8][16][32];
  __shared__ __align__(16) unsigned short pb[8][16][32];
  __shared__ float rsc[8][16];
  __shared__ float rl[8][16];

  int wv = threadIdx.x >> 5, lane = threadIdx.x & 31;
  int n = lane & 15, hh = lane >> 4;
  int g = blockIdx.x * 8 + wv;
  const int QT = T_ / 16;
  int qt = g % QT;
  int head = (g / QT) % H_;
  int b = g / (QT * H_);
  int qBase = qt * 16;

  const unsigned short* qb = q  + (size_t)(b * H_ + head) * T_ * D_;
  const unsigned short* kb = k  + (size_t)(b * H_ + head) * T_ * D_;
  const unsigned short* vb = vt + (size_t)(b * H_ + head) * D_ * T_;

  v8f accO[4] = {};
  float mrow = -INFINITY, lrow = 0.f;   // live on lanes 0..15 only
  const float scale = 0.03125f;          // C_**-0.5 = 1/32

  int sEnd = qBase + 16;                 // causal bound
  for (int s0 = 0; s0 < sEnd; s0 += 32) {
    // ---- scores S = Q K^T for 16x32 key block (2 n-tiles, 2 k-steps each)
#pragma unroll
    for (int jt = 0; jt < 2; ++jt) {
      v8f accS = {};
#pragma unroll
      for (int kk = 0; kk < D_; kk += 32) {
        Frag a, bfrag;
        const unsigned short* ar = qb + (size_t)(qBase + n) * D_ + kk + hh * 16;
        a.q[0] = *(const uint4*)ar;
        a.q[1] = *(const uint4*)(ar + 8);
        const unsigned short* br =
            kb + (size_t)(s0 + jt * 16 + n) * D_ + kk + hh * 16;
        bfrag.q[0] = ((const uint4*)br)[0];
        bfrag.q[1] = ((const uint4*)br)[1];
        accS = __builtin_amdgcn_wmma_f32_16x16x32_bf16(
            false, a.v, false, bfrag.v, (short)0, accS, false, false);
      }
#pragma unroll
      for (int i = 0; i < 8; ++i)
        sc[wv][i + 8 * hh][jt * 16 + n] = accS[i] * scale;
    }
    asm volatile("s_wait_dscnt 0x0" ::: "memory");

    // ---- online softmax row pass (lanes 0..15, one row each)
    if (lane < 16) {
      int r = lane, t = qBase + r;
      float mblk = -INFINITY;
      for (int c = 0; c < 32; ++c) {
        int s = s0 + c;
        bool valid = (s <= t) && (kpm[(size_t)b * T_ + s] != 0);
        float vv = valid ? sc[wv][r][c] : -INFINITY;
        mblk = fmaxf(mblk, vv);
      }
      float mnew = fmaxf(mrow, mblk);
      float scl = (mrow == -INFINITY) ? 0.f : __expf(mrow - mnew);
      float psum = 0.f;
      for (int c = 0; c < 32; ++c) {
        int s = s0 + c;
        bool valid = (s <= t) && (kpm[(size_t)b * T_ + s] != 0);
        float vv = valid ? sc[wv][r][c] : -INFINITY;
        float p = (vv == -INFINITY) ? 0.f : __expf(vv - mnew);
        pb[wv][r][c] = f2bf(p);
        psum += p;
      }
      lrow = lrow * scl + psum;
      mrow = mnew;
      rsc[wv][r] = scl;
      rl[wv][r] = lrow;
    }
    asm volatile("s_wait_dscnt 0x0" ::: "memory");

    // ---- rescale running output accumulators
#pragma unroll
    for (int i = 0; i < 8; ++i) {
      float f = rsc[wv][i + 8 * hh];
#pragma unroll
      for (int j = 0; j < 4; ++j) accO[j][i] *= f;
    }

    // ---- O += P @ V  (A-frag of P from LDS; Bt = V^T rows contiguous in s)
    Frag pa;
    const unsigned short* pr = &pb[wv][n][0];
    pa.q[0] = *(const uint4*)(pr + hh * 16);
    pa.q[1] = *(const uint4*)(pr + 8 + hh * 16);
#pragma unroll
    for (int j = 0; j < 4; ++j) {
      Frag bfrag;
      const unsigned short* br = vb + (size_t)(j * 16 + n) * T_ + s0 + hh * 16;
      bfrag.q[0] = ((const uint4*)br)[0];
      bfrag.q[1] = ((const uint4*)br)[1];
      accO[j] = __builtin_amdgcn_wmma_f32_16x16x32_bf16(
          false, pa.v, false, bfrag.v, (short)0, accO[j], false, false);
    }
  }

  asm volatile("s_wait_dscnt 0x0" ::: "memory");
  // ---- epilogue: divide by l, scatter bf16 to [B,T,C]
#pragma unroll
  for (int j = 0; j < 4; ++j) {
#pragma unroll
    for (int i = 0; i < 8; ++i) {
      int r = i + 8 * hh;
      float o = accO[j][i] / rl[wv][r];
      int t = qBase + r;
      attn_out[(size_t)(b * T_ + t) * C_ + head * D_ + j * 16 + n] = f2bf(o);
    }
  }
}

// ---------------------------------------------------------------------------
extern "C" void kernel_launch(void* const* d_in, const int* in_sizes, int n_in,
                              void* d_out, int out_size, void* d_ws, size_t ws_size,
                              hipStream_t stream) {
  const float* x       = (const float*)d_in[0];
  const unsigned char* kpm = (const unsigned char*)d_in[1];
  const float* wq      = (const float*)d_in[2];
  const float* wk      = (const float*)d_in[3];
  const float* wv      = (const float*)d_in[4];
  const float* proj_w  = (const float*)d_in[5];
  const float* proj_b  = (const float*)d_in[6];
  const float* ff_w1   = (const float*)d_in[7];
  const float* ff_b1   = (const float*)d_in[8];
  const float* ff_w2   = (const float*)d_in[9];
  const float* ff_b2   = (const float*)d_in[10];
  const float* ln1_w   = (const float*)d_in[11];
  const float* ln1_b   = (const float*)d_in[12];
  const float* ln2_w   = (const float*)d_in[13];
  const float* ln2_b   = (const float*)d_in[14];
  float* out = (float*)d_out;

  // --- workspace carve-up (bytes) ---
  char* p = (char*)d_ws;
  unsigned short* h_bf   = (unsigned short*)p; p += (size_t)BT_ * C_ * 2;
  unsigned short* q_bf   = (unsigned short*)p; p += (size_t)BT_ * C_ * 2;
  unsigned short* k_bf   = (unsigned short*)p; p += (size_t)BT_ * C_ * 2;
  unsigned short* vt_bf  = (unsigned short*)p; p += (size_t)BT_ * C_ * 2;
  unsigned short* at_bf  = (unsigned short*)p; p += (size_t)BT_ * C_ * 2;
  unsigned short* h2_bf  = (unsigned short*)p; p += (size_t)BT_ * C_ * 2;
  unsigned short* ff1_bf = (unsigned short*)p; p += (size_t)BT_ * DFF_ * 2;
  float*          x1     = (float*)p;          p += (size_t)BT_ * C_ * 4;
  unsigned short* wqT    = (unsigned short*)p; p += (size_t)C_ * C_ * 2;
  unsigned short* wkT    = (unsigned short*)p; p += (size_t)C_ * C_ * 2;
  unsigned short* wvT    = (unsigned short*)p; p += (size_t)C_ * C_ * 2;
  unsigned short* pwT    = (unsigned short*)p; p += (size_t)C_ * C_ * 2;
  unsigned short* f1T    = (unsigned short*)p; p += (size_t)C_ * DFF_ * 2;
  unsigned short* f2T    = (unsigned short*)p; p += (size_t)DFF_ * C_ * 2;

  // 1) weight transpose+convert to bf16 [N,K]
  int nCC  = C_ * C_;
  int nCF  = C_ * DFF_;
  transpose_bf16_kernel<<<(nCC + 255) / 256, 256, 0, stream>>>(wq, wqT, C_, C_);
  transpose_bf16_kernel<<<(nCC + 255) / 256, 256, 0, stream>>>(wk, wkT, C_, C_);
  transpose_bf16_kernel<<<(nCC + 255) / 256, 256, 0, stream>>>(wv, wvT, C_, C_);
  transpose_bf16_kernel<<<(nCC + 255) / 256, 256, 0, stream>>>(proj_w, pwT, C_, C_);
  transpose_bf16_kernel<<<(nCF + 255) / 256, 256, 0, stream>>>(ff_w1, f1T, C_, DFF_);
  transpose_bf16_kernel<<<(nCF + 255) / 256, 256, 0, stream>>>(ff_w2, f2T, DFF_, C_);

  // 2) h = LN1(x)*mask  (bf16)
  ln_kernel<<<BT_, 256, 0, stream>>>(x, ln1_w, ln1_b, kpm, h_bf);

  // 3-5) Q,K,V projections (WMMA), scattered into attention layouts
  dim3 gQKV(C_ / 64, BT_ / 128);
  gemm_bf16_kernel<M_QK><<<gQKV, 256, 0, stream>>>(h_bf, wqT, BT_, C_, C_,
      nullptr, q_bf, nullptr, nullptr, nullptr);
  gemm_bf16_kernel<M_QK><<<gQKV, 256, 0, stream>>>(h_bf, wkT, BT_, C_, C_,
      nullptr, k_bf, nullptr, nullptr, nullptr);
  gemm_bf16_kernel<M_VT><<<gQKV, 256, 0, stream>>>(h_bf, wvT, BT_, C_, C_,
      nullptr, vt_bf, nullptr, nullptr, nullptr);

  // 6) flash attention
  int nWaves = B_ * H_ * (T_ / 16);
  attn_kernel<<<nWaves / 8, 256, 0, stream>>>(q_bf, k_bf, vt_bf, kpm, at_bf);

  // 7) x1 = x + attn @ proj_w + proj_b
  gemm_bf16_kernel<M_PROJ><<<gQKV, 256, 0, stream>>>(at_bf, pwT, BT_, C_, C_,
      x1, nullptr, proj_b, x, nullptr);

  // 8) h2 = LN2(x1)*mask
  ln_kernel<<<BT_, 256, 0, stream>>>(x1, ln2_w, ln2_b, kpm, h2_bf);

  // 9) ff1 = relu(h2 @ ff_w1 + ff_b1)  (bf16)
  dim3 gF1(DFF_ / 64, BT_ / 128);
  gemm_bf16_kernel<M_RELU><<<gF1, 256, 0, stream>>>(h2_bf, f1T, BT_, DFF_, C_,
      nullptr, ff1_bf, ff_b1, nullptr, nullptr);

  // 10) out = (x1 + ff1 @ ff_w2 + ff_b2) * mask
  dim3 gF2(C_ / 64, BT_ / 128);
  gemm_bf16_kernel<M_FF2><<<gF2, 256, 0, stream>>>(ff1_bf, f2T, BT_, C_, DFF_,
      out, nullptr, ff_b2, x1, kpm);
}
